// GCNDecoder_34359739015
// MI455X (gfx1250) — matrix operand order repaired and checked by reading, not compiled, not run on hardware
//
#include <hip/hip_runtime.h>

// ---------------------------------------------------------------------------
// GCN edge decoder for MI455X (gfx1250, wave32, WMMA).
//
// Pipeline (all on `stream`, BatchNorm forces per-layer splits):
//   prep:  W4/W1/W2 (f32, [K][N]) -> Wt (f16, [N][K]) for B-fragment loads
//   L4:    zero stats; GEMM(gather(x,eidx) -> H=edge_x@W4, col sums);
//          finalize a,c; act: act4 = f16(relu(H*a+c))
//   L1:    zero; GEMM(act4@W1 -> H, sums); finalize; out1 = f16(relu(..)+act4)
//   L2:    zero; GEMM(out1@W2 -> H, sums); finalize;
//   final: y[e] = sum_n (relu(H*a+c)+out1)[n] * Wf[n] + bf
//
// Workspace: Wt4|Wt1|Wt2 (384KB) | sums/sq/a/c (4KB) | H f32 (E*256*4, reused
// for all 3 layers) | act4 f16 | out1 f16  => ~1.0 GB total.
// Linear biases cancel inside BatchNorm and are skipped.
// ---------------------------------------------------------------------------

typedef __attribute__((ext_vector_type(16))) _Float16 v16h;
typedef __attribute__((ext_vector_type(8)))  float    v8f;
typedef int v4i __attribute__((vector_size(16)));

#define D_DIM 256
#define TILE_M 64
#define THREADS_GEMM 512

#if __has_builtin(__builtin_amdgcn_global_load_async_to_lds_b128)
#define HAVE_ASYNC_LDS 1
#else
#define HAVE_ASYNC_LDS 0
#endif

// A fragment (16-bit A 16x32 layout): per lane take k[0..7] and k[16..23]
__device__ __forceinline__ v16h load_a_frag(const _Float16* p) {
  union { uint4 u[2]; v16h v; } f;
  f.u[0] = *(const uint4*)(p);
  f.u[1] = *(const uint4*)(p + 16);
  return f.v;
}
// B fragment: per lane take k[0..7] and k[16..23] (same order as A)
__device__ __forceinline__ v16h load_b_frag(const _Float16* p) {
  union { uint4 u[2]; v16h v; } f;
  f.u[0] = *(const uint4*)(p);
  f.u[1] = *(const uint4*)(p + 16);
  return f.v;
}

#if HAVE_ASYNC_LDS
// async DMA 16B global -> LDS (ASYNCcnt); OFF applies to both addresses.
// Builtin signature (from hipcc diagnostic): (int4 AS1*, int4 AS3*, imm, imm)
template<int OFF>
__device__ __forceinline__ void async_copy16(const void* g, void* l) {
  __builtin_amdgcn_global_load_async_to_lds_b128(
      (__attribute__((address_space(1))) v4i*)g,
      (__attribute__((address_space(3))) v4i*)l, OFF, 0);
}
#endif

// C = A[64xD] @ W[DxD] for a tile of 64 rows; writes H (f32) and accumulates
// per-column sum / sum-of-squares into gsum/gsq (global atomics, block-reduced).
// GATHER=1: A rows are concat(x[src], x[dst]) converted f32->f16 on the fly.
template<int GATHER>
__global__ __launch_bounds__(THREADS_GEMM)
void gemm_stats(const _Float16* __restrict__ Af16,
                const float*    __restrict__ x,
                const int*      __restrict__ eidx,
                const _Float16* __restrict__ Wt,   // [N][K] f16
                float*          __restrict__ H,    // [E][D] f32
                float*          __restrict__ gsum,
                float*          __restrict__ gsq,
                int E, int F)
{
  __shared__ __align__(16) _Float16 As[TILE_M * D_DIM];
  __shared__ float ssum[D_DIM];
  __shared__ float ssq[D_DIM];

  const int tid   = threadIdx.x;
  const int wave  = tid >> 5;          // 16 waves, one 16-wide N tile each
  const int lane  = tid & 31;
  const int lhalf = lane >> 4;
  const int l16   = lane & 15;
  const int ncol  = wave * 16 + l16;   // output column owned by this lane

  // B fragments held in VGPRs for the whole kernel (8 x v16h = 64 VGPRs)
  v16h bfrag[8];
  {
    const _Float16* wrow = Wt + (size_t)ncol * D_DIM;
    #pragma unroll
    for (int ks = 0; ks < 8; ++ks)
      bfrag[ks] = load_b_frag(wrow + ks * 32 + lhalf * 8);
  }

  if (tid < D_DIM) { ssum[tid] = 0.f; ssq[tid] = 0.f; }
  __syncthreads();

  float lsum = 0.f, lsq = 0.f;

  const int ntiles = (E + TILE_M - 1) / TILE_M;
  const int frow = tid >> 3;           // staging: 8 threads per row
  const int fcol = (tid & 7) * 32;     // 32 contiguous cols per thread

  for (int t = blockIdx.x; t < ntiles; t += gridDim.x) {
    const int e0   = t * TILE_M;
    const int cnt  = min(TILE_M, E - e0);
    const bool full = (cnt == TILE_M);   // uniform across block

    __syncthreads();                   // previous tile's As reads done
    // ---- stage A tile (64 x 256 f16) into LDS ----
    if (GATHER) {
      if (frow < cnt) {
        const int e    = e0 + frow;
        const int node = (fcol < F) ? eidx[e] : eidx[E + e];
        const int coff = (fcol < F) ? fcol : (fcol - F);
        const float* sp = x + (size_t)node * F + coff;
        #pragma unroll
        for (int i = 0; i < 32; ++i)
          As[frow * D_DIM + fcol + i] = (_Float16)sp[i];
      } else {
        #pragma unroll
        for (int i = 0; i < 32; ++i)
          As[frow * D_DIM + fcol + i] = (_Float16)0.f;
      }
    } else {
#if HAVE_ASYNC_LDS
      if (frow < cnt) {
        const _Float16* sp = Af16 + (size_t)(e0 + frow) * D_DIM + fcol;
        _Float16* lp = &As[frow * D_DIM + fcol];
        async_copy16<0>(sp, lp);
        async_copy16<16>(sp, lp);
        async_copy16<32>(sp, lp);
        async_copy16<48>(sp, lp);
      } else {
        #pragma unroll
        for (int i = 0; i < 32; ++i)
          As[frow * D_DIM + fcol + i] = (_Float16)0.f;
      }
      asm volatile("s_wait_asynccnt 0x0" ::: "memory");
#else
      if (frow < cnt) {
        const _Float16* sp = Af16 + (size_t)(e0 + frow) * D_DIM + fcol;
        #pragma unroll
        for (int i = 0; i < 32; ++i)
          As[frow * D_DIM + fcol + i] = sp[i];
      } else {
        #pragma unroll
        for (int i = 0; i < 32; ++i)
          As[frow * D_DIM + fcol + i] = (_Float16)0.f;
      }
#endif
    }
    __syncthreads();

    #pragma unroll
    for (int mt = 0; mt < 4; ++mt) {   // 4 M-tiles of 16 edges
      const _Float16* arow = &As[(mt * 16 + l16) * D_DIM];
      // prefetch all 8 A fragments so ds_loads batch ahead of the WMMA chain
      v16h afrag[8];
      #pragma unroll
      for (int ks = 0; ks < 8; ++ks)
        afrag[ks] = load_a_frag(arow + ks * 32 + lhalf * 8);

      v8f c = {};
      #pragma unroll
      for (int ks = 0; ks < 8; ++ks)
        c = __builtin_amdgcn_wmma_f32_16x16x32_f16(false, afrag[ks], false,
                                                   bfrag[ks], (short)0, c,
                                                   false, false);

      const int mbase = mt * 16 + lhalf * 8;   // C layout: VGPR j -> M=j(+8)
      if (full) {
        // one base address, 8 stores with immediate offsets (j*1024 bytes)
        float* hp = H + (size_t)(e0 + mbase) * D_DIM + ncol;
        #pragma unroll
        for (int j = 0; j < 8; ++j) {
          const float v = c[j];
          lsum += v; lsq += v * v;
          hp[j * D_DIM] = v;
        }
      } else {
        #pragma unroll
        for (int j = 0; j < 8; ++j) {
          const float v = c[j];
          lsum += v; lsq += v * v;          // padded rows contribute exact 0
          const int mrow = mbase + j;
          if (mrow < cnt)
            H[(size_t)(e0 + mrow) * D_DIM + ncol] = v;
        }
      }
    }
  }

  __syncthreads();
  atomicAdd(&ssum[ncol], lsum);
  atomicAdd(&ssq[ncol],  lsq);
  __syncthreads();
  if (tid < D_DIM) {
    atomicAdd(&gsum[tid], ssum[tid]);
    atomicAdd(&gsq[tid],  ssq[tid]);
  }
}

// (sum, sumsq) -> affine (a, c): bn(h) = h*a + c
__global__ void finalize_scales(const float* __restrict__ gsum,
                                const float* __restrict__ gsq,
                                const float* __restrict__ g,
                                const float* __restrict__ be,
                                float* __restrict__ a, float* __restrict__ c,
                                float invE)
{
  const int n = threadIdx.x;
  const float m = gsum[n] * invE;
  const float v = gsq[n] * invE - m * m;
  const float s = g[n] * rsqrtf(v + 1e-5f);
  a[n] = s;
  c[n] = be[n] - m * s;
}

// out = f16(relu(H*a + c) [+ resid])
__global__ void act_pass(const float* __restrict__ H,
                         const float* __restrict__ a,
                         const float* __restrict__ c,
                         const _Float16* __restrict__ resid,
                         _Float16* __restrict__ outp,
                         size_t total)
{
  const size_t stride = (size_t)gridDim.x * blockDim.x * 8;
  for (size_t i = ((size_t)blockIdx.x * blockDim.x + threadIdx.x) * 8;
       i < total; i += stride) {
    const int n = (int)(i & (D_DIM - 1));
    #pragma unroll
    for (int j = 0; j < 8; ++j) {
      float v = H[i + j] * a[n + j] + c[n + j];
      v = v > 0.f ? v : 0.f;
      if (resid) v += (float)resid[i + j];
      outp[i + j] = (_Float16)v;
    }
  }
}

// y[e] = sum_n (relu(H*a+c) + out1)[n] * Wf[n] + bf ; 8 threads per edge
__global__ __launch_bounds__(256)
void final_pass(const float* __restrict__ H,
                const _Float16* __restrict__ out1,
                const float* __restrict__ a,
                const float* __restrict__ c,
                const float* __restrict__ Wf,
                const float* __restrict__ bf,
                float* __restrict__ y, int E)
{
  __shared__ float a_s[D_DIM], c_s[D_DIM], w_s[D_DIM], red[256];
  const int tid = threadIdx.x;
  a_s[tid] = a[tid]; c_s[tid] = c[tid]; w_s[tid] = Wf[tid];
  __syncthreads();

  const int el = tid >> 3, part = tid & 7;
  const int ngroups = (E + 31) / 32;
  for (int gr = blockIdx.x; gr < ngroups; gr += gridDim.x) {
    const int e = gr * 32 + el;
    float acc = 0.f;
    if (e < E) {
      const float*    hrow = H    + (size_t)e * D_DIM;
      const _Float16* orow = out1 + (size_t)e * D_DIM;
      const int cb = part * 32;
      #pragma unroll 8
      for (int i = 0; i < 32; ++i) {
        const int n = cb + i;
        float v = hrow[n] * a_s[n] + c_s[n];
        v = v > 0.f ? v : 0.f;
        v += (float)orow[n];
        acc += v * w_s[n];
      }
    }
    red[tid] = acc;
    __syncthreads();
    if (part == 0 && e < E) {
      float s = red[el * 8];
      #pragma unroll
      for (int k = 1; k < 8; ++k) s += red[el * 8 + k];
      y[e] = s + bf[0];
    }
    __syncthreads();
  }
}

// W f32 [K][N] -> Wt f16 [N][K]
__global__ void prep_wt(const float* __restrict__ W, _Float16* __restrict__ Wt,
                        int D)
{
  const int idx = blockIdx.x * blockDim.x + threadIdx.x;
  if (idx < D * D) {
    const int n = idx / D, k = idx % D;
    Wt[(size_t)n * D + k] = (_Float16)W[(size_t)k * D + n];
  }
}

__global__ void zero_buf(float* __restrict__ p, int n)
{
  const int i = blockIdx.x * blockDim.x + threadIdx.x;
  if (i < n) p[i] = 0.f;
}

extern "C" void kernel_launch(void* const* d_in, const int* in_sizes, int n_in,
                              void* d_out, int out_size, void* d_ws, size_t ws_size,
                              hipStream_t stream)
{
  const float* x    = (const float*)d_in[0];
  const int*   eidx = (const int*)  d_in[1];   // (2,E) int32
  const float* W4   = (const float*)d_in[2];
  const float* g4   = (const float*)d_in[4];
  const float* be4  = (const float*)d_in[5];
  const float* W1   = (const float*)d_in[6];
  const float* g1   = (const float*)d_in[8];
  const float* be1  = (const float*)d_in[9];
  const float* W2   = (const float*)d_in[10];
  const float* g2   = (const float*)d_in[12];
  const float* be2  = (const float*)d_in[13];
  const float* Wf   = (const float*)d_in[14];
  const float* bf   = (const float*)d_in[15];
  float* y = (float*)d_out;

  const int E = in_sizes[1] / 2;
  int D = 1; while ((long long)D * D < (long long)in_sizes[2]) D++;  // 256
  const int F = D / 2;                                               // 128

  // ---- workspace carve-up ----
  char* ws = (char*)d_ws;
  _Float16* Wt4 = (_Float16*)ws;
  _Float16* Wt1 = Wt4 + (size_t)D * D;
  _Float16* Wt2 = Wt1 + (size_t)D * D;
  size_t off = 3 * (size_t)D * D * sizeof(_Float16);
  float* ssum   = (float*)(ws + off);            off += (size_t)D * 4;
  float* ssq    = (float*)(ws + off);            off += (size_t)D * 4;
  float* ascale = (float*)(ws + off);            off += (size_t)D * 4;
  float* cshift = (float*)(ws + off);            off += (size_t)D * 4;
  off = (off + 255) & ~(size_t)255;
  float* H = (float*)(ws + off);                 off += (size_t)E * D * 4;
  _Float16* act4 = (_Float16*)(ws + off);        off += (size_t)E * D * 2;
  _Float16* out1 = (_Float16*)(ws + off);

  const size_t total = (size_t)E * D;
  const int ntiles  = (E + TILE_M - 1) / TILE_M;
  const int gblocks = ntiles < 2048 ? ntiles : 2048;
  const int ablocks = (int)(((total / 8) + 255) / 256) < 8192
                    ? (int)(((total / 8) + 255) / 256) : 8192;
  const int fgroups = (E + 31) / 32;
  const int fblocks = fgroups < 4096 ? fgroups : 4096;
  const int pblocks = (D * D + 255) / 256;
  const float invE  = 1.0f / (float)E;

  prep_wt<<<pblocks, 256, 0, stream>>>(W4, Wt4, D);
  prep_wt<<<pblocks, 256, 0, stream>>>(W1, Wt1, D);
  prep_wt<<<pblocks, 256, 0, stream>>>(W2, Wt2, D);

  // ---- block 4: gather -> Linear -> BN -> ReLU ----
  zero_buf<<<(2 * D + 255) / 256, 256, 0, stream>>>(ssum, 2 * D);
  gemm_stats<1><<<gblocks, THREADS_GEMM, 0, stream>>>(
      nullptr, x, eidx, Wt4, H, ssum, ssq, E, F);
  finalize_scales<<<1, D, 0, stream>>>(ssum, ssq, g4, be4, ascale, cshift, invE);
  act_pass<<<ablocks, 256, 0, stream>>>(H, ascale, cshift, nullptr, act4, total);

  // ---- block 1: Linear -> BN -> ReLU + residual ----
  zero_buf<<<(2 * D + 255) / 256, 256, 0, stream>>>(ssum, 2 * D);
  gemm_stats<0><<<gblocks, THREADS_GEMM, 0, stream>>>(
      act4, nullptr, nullptr, Wt1, H, ssum, ssq, E, F);
  finalize_scales<<<1, D, 0, stream>>>(ssum, ssq, g1, be1, ascale, cshift, invE);
  act_pass<<<ablocks, 256, 0, stream>>>(H, ascale, cshift, act4, out1, total);

  // ---- block 2 + final projection ----
  zero_buf<<<(2 * D + 255) / 256, 256, 0, stream>>>(ssum, 2 * D);
  gemm_stats<0><<<gblocks, THREADS_GEMM, 0, stream>>>(
      out1, nullptr, nullptr, Wt2, H, ssum, ssq, E, F);
  finalize_scales<<<1, D, 0, stream>>>(ssum, ssq, g2, be2, ascale, cshift, invE);
  final_pass<<<fblocks, 256, 0, stream>>>(H, out1, ascale, cshift, Wf, bf, y, E);
}
